// IMLA_54262616817908
// MI455X (gfx1250) — compile-verified, matrix operand order checked by reading, not run on hardware
//
#include <hip/hip_runtime.h>
#include <hip/hip_bf16.h>

// ---- problem constants (B=1) ----
#define T_   2048
#define D_   2048
#define NH_  16
#define DH_  128
#define DRH_ 64
#define DC_  512
#define QKD_ 192   // DH + DRH

typedef __attribute__((ext_vector_type(16))) __bf16 v16bf;
typedef __attribute__((ext_vector_type(8)))  float  v8f;
typedef int v4i_vs __attribute__((vector_size(16)));   // matches builtin param

// ---- optional CDNA5 async copy (memory -> LDS, ASYNCcnt tracked) ----
#if defined(__has_builtin)
#  if __has_builtin(__builtin_amdgcn_global_load_async_to_lds_b128)
#    define HAVE_ASYNC_COPY 1
#  endif
#  if __has_builtin(__builtin_amdgcn_s_wait_asynccnt)
#    define HAVE_WAIT_ASYNC 1
#  endif
#endif
#ifndef HAVE_ASYNC_COPY
#  define HAVE_ASYNC_COPY 0
#endif
#ifndef HAVE_WAIT_ASYNC
#  define HAVE_WAIT_ASYNC 0
#endif

__device__ __forceinline__ void cp16_g2l(void* l, const void* g) {
#if HAVE_ASYNC_COPY
  __builtin_amdgcn_global_load_async_to_lds_b128(
      (__attribute__((address_space(1))) v4i_vs*)g,
      (__attribute__((address_space(3))) v4i_vs*)l, 0, 0);
#else
  *(uint4*)l = *(const uint4*)g;
#endif
}
__device__ __forceinline__ void async_copy_wait() {
#if HAVE_ASYNC_COPY
#  if HAVE_WAIT_ASYNC
  __builtin_amdgcn_s_wait_asynccnt(0);
#  else
  asm volatile("s_wait_asynccnt 0x0" ::: "memory");
#  endif
#endif
}

// ---- bf16 helpers ----
__device__ __forceinline__ unsigned short f2bf_u16(float f) {
  unsigned int u = __builtin_bit_cast(unsigned int, f);
  unsigned int r = u + 0x7fffu + ((u >> 16) & 1u);   // round to nearest even
  return (unsigned short)(r >> 16);
}
__device__ __forceinline__ __bf16 u2bf(unsigned short u) {
  return __builtin_bit_cast(__bf16, u);
}
__device__ __forceinline__ float bf2f(unsigned short u) {
  unsigned int v = ((unsigned int)u) << 16;
  return __builtin_bit_cast(float, v);
}
// 16-bit WMMA A/B fragment K index for (lane, half-slot j):
// lanes 0-15 hold K{0..7,16..23}; lanes 16-31 hold K{8..15,24..31}
__device__ __forceinline__ int frag_k(int lane, int j) {
  return ((lane >> 4) << 3) + (j & 7) + ((j & 8) ? 16 : 0);
}

// ============================================================================
// fp32 -> bf16 pack (vectorized, one-time producers)
// ============================================================================
__global__ void pack_bf16_kernel(const float* __restrict__ in,
                                 unsigned short* __restrict__ out, int n4)
{
  int idx = blockIdx.x * blockDim.x + threadIdx.x;
  if (idx >= n4) return;
  float4 f = ((const float4*)in)[idx];
  ushort4 u;
  u.x = f2bf_u16(f.x); u.y = f2bf_u16(f.y);
  u.z = f2bf_u16(f.z); u.w = f2bf_u16(f.w);
  ((ushort4*)out)[idx] = u;
}

// ============================================================================
// Generic batched GEMM: C[M,N] = A[M,K] * B[N,K]^T, A/B bf16 in memory,
// bf16 WMMA compute, fp32 accumulate, out fp32 or bf16.
// 64x64 block tile, 128 threads (4 waves), each wave = 16 rows x 64 cols.
// Staging is pure 16-byte copies (async-to-LDS when available).
// ============================================================================
__global__ __launch_bounds__(128) void wmma_gemm_bf16(
    const unsigned short* __restrict__ A, int lda, long long strideA,
    const unsigned short* __restrict__ B, int ldb, long long strideB,
    void* __restrict__ C, int ldc, long long strideC,
    int K, int bf16_out)
{
  __shared__ __attribute__((aligned(16))) unsigned short as[64 * 32];
  __shared__ __attribute__((aligned(16))) unsigned short bs[64 * 32];
  const int tid  = threadIdx.x;
  const int wave = tid >> 5, lane = tid & 31;
  const int hf = lane >> 4, ln = lane & 15;
  A += (long long)blockIdx.z * strideA;
  B += (long long)blockIdx.z * strideB;
  const int row0 = blockIdx.y * 64;
  const int col0 = blockIdx.x * 64;

  v8f acc[4];
  #pragma unroll
  for (int c = 0; c < 4; ++c)
    #pragma unroll
    for (int e = 0; e < 8; ++e) acc[c][e] = 0.0f;

  for (int k0 = 0; k0 < K; k0 += 32) {
    // 64x32 bf16 tile = 256 x 16B chunks; 2 chunks per thread per operand
    #pragma unroll
    for (int i = 0; i < 2; ++i) {
      int idx = tid + i * 128;
      int r = idx >> 2, c8 = (idx & 3) * 8;
      cp16_g2l(&as[r * 32 + c8], A + (long long)(row0 + r) * lda + k0 + c8);
      cp16_g2l(&bs[r * 32 + c8], B + (long long)(col0 + r) * ldb + k0 + c8);
    }
    async_copy_wait();
    __syncthreads();

    v16bf a;
    #pragma unroll
    for (int j = 0; j < 16; ++j)
      a[j] = u2bf(as[(wave * 16 + ln) * 32 + frag_k(lane, j)]);

    #pragma unroll
    for (int c = 0; c < 4; ++c) {
      v16bf b;
      #pragma unroll
      for (int j = 0; j < 16; ++j)
        b[j] = u2bf(bs[(c * 16 + ln) * 32 + frag_k(lane, j)]);
      acc[c] = __builtin_amdgcn_wmma_f32_16x16x32_bf16(
          false, a, false, b, (short)0, acc[c], false, false);
    }
    __syncthreads();
  }

  if (bf16_out) {
    unsigned short* Cb = (unsigned short*)C + (long long)blockIdx.z * strideC;
    #pragma unroll
    for (int c = 0; c < 4; ++c)
      #pragma unroll
      for (int r = 0; r < 8; ++r)
        Cb[(long long)(row0 + wave * 16 + r + 8 * hf) * ldc + col0 + c * 16 + ln] =
            f2bf_u16(acc[c][r]);
  } else {
    float* Cf = (float*)C + (long long)blockIdx.z * strideC;
    #pragma unroll
    for (int c = 0; c < 4; ++c)
      #pragma unroll
      for (int r = 0; r < 8; ++r)
        Cf[(long long)(row0 + wave * 16 + r + 8 * hf) * ldc + col0 + c * 16 + ln] =
            acc[c][r];
  }
}

// ============================================================================
// RoPE on k_r (fp32 in); emits ck cache (bf16 -> d_out) and broadcasts
// bf16(rope/NH) into each head's K-buffer columns [128..191].
// ============================================================================
__global__ void rope_k_kernel(const float* __restrict__ kr,
                              const float* __restrict__ fc,
                              const float* __restrict__ fs,
                              __hip_bfloat16* __restrict__ ck_out,
                              unsigned short* __restrict__ Kb)
{
  int idx = blockIdx.x * blockDim.x + threadIdx.x;   // T * (DRH/2)
  if (idx >= T_ * 32) return;
  int t = idx >> 5, i = idx & 31;
  float xr = kr[t * DRH_ + 2 * i], xi = kr[t * DRH_ + 2 * i + 1];
  float c = fc[t * 32 + i], s = fs[t * 32 + i];
  float o0 = xr * c - xi * s;
  float o1 = xr * s + xi * c;
  unsigned short b0 = f2bf_u16(o0), b1 = f2bf_u16(o1);
  ck_out[t * DRH_ + 2 * i]     = __float2bfloat16(o0);
  ck_out[t * DRH_ + 2 * i + 1] = __float2bfloat16(o1);
  // bf16(x)/16 is exact in bf16 (exponent shift)
  unsigned short k0 = f2bf_u16(bf2f(b0) * (1.0f / NH_));
  unsigned short k1 = f2bf_u16(bf2f(b1) * (1.0f / NH_));
  #pragma unroll
  for (int n = 0; n < NH_; ++n) {
    long long base = ((long long)n * T_ + t) * QKD_ + DH_;
    Kb[base + 2 * i]     = k0;
    Kb[base + 2 * i + 1] = k1;
  }
}

// RoPE on q_r in place inside the bf16 Q buffer columns [128..191].
__global__ void rope_q_kernel(unsigned short* __restrict__ Qb,
                              const float* __restrict__ fc,
                              const float* __restrict__ fs)
{
  int idx = blockIdx.x * blockDim.x + threadIdx.x;   // NH * T * 32
  if (idx >= NH_ * T_ * 32) return;
  int n = idx >> 16;           // T*32 = 65536 per head
  int rem = idx & 65535;
  int t = rem >> 5, i = rem & 31;
  long long base = ((long long)n * T_ + t) * QKD_ + DH_ + 2 * i;
  float xr = bf2f(Qb[base]), xi = bf2f(Qb[base + 1]);
  float c = fc[t * 32 + i], s = fs[t * 32 + i];
  Qb[base]     = f2bf_u16(xr * c - xi * s);
  Qb[base + 1] = f2bf_u16(xr * s + xi * c);
}

// ============================================================================
// Flash attention per (64-row query tile, head). 4 waves, each owns a 16-row
// strip. Q/K/V bf16, causal, scale 1/sqrt(192), online softmax.
// Output written bf16 as O[t, h'*NH + head] for the final projection.
// ============================================================================
__global__ __launch_bounds__(128) void mla_attn_kernel(
    const unsigned short* __restrict__ Qb, const unsigned short* __restrict__ Kb,
    const unsigned short* __restrict__ Vb, unsigned short* __restrict__ Ob)
{
  __shared__ __attribute__((aligned(16))) unsigned short ks[64 * QKD_];   // 24 KB
  __shared__ __attribute__((aligned(16))) unsigned short vs[64 * DH_];    // 16 KB
  __shared__ __attribute__((aligned(16))) unsigned short ps[4 * 16 * 64]; //  8 KB

  const int tid = threadIdx.x, wave = tid >> 5, lane = tid & 31;
  const int hf = lane >> 4, ln = lane & 15;
  const int head = blockIdx.y;
  const int q0 = blockIdx.x * 64;
  const int qrow = q0 + wave * 16;

  // preload Q fragments (6 k-chunks of 32 over 192); 8-half runs are
  // contiguous -> merges to global_load_b128
  v16bf qf[6];
  const unsigned short* Qrow = Qb + ((long long)head * T_ + qrow + ln) * QKD_;
  #pragma unroll
  for (int kk = 0; kk < 6; ++kk)
    #pragma unroll
    for (int j = 0; j < 16; ++j)
      qf[kk][j] = u2bf(Qrow[kk * 32 + frag_k(lane, j)]);

  v8f o[8];
  #pragma unroll
  for (int nt = 0; nt < 8; ++nt)
    #pragma unroll
    for (int e = 0; e < 8; ++e) o[nt][e] = 0.0f;
  float mrun[8], lrun[8];
  #pragma unroll
  for (int r = 0; r < 8; ++r) { mrun[r] = -3.0e38f; lrun[r] = 0.0f; }

  const float scl = 0.07216878364870323f;   // 1/sqrt(192)
  const int nkb = blockIdx.x + 1;           // causal: key tiles <= query tile
  for (int kb = 0; kb < nkb; ++kb) {
    const int kbase = kb * 64;
    const unsigned short* Ksrc = Kb + ((long long)head * T_ + kbase) * QKD_;
    const unsigned short* Vsrc = Vb + ((long long)head * T_ + kbase) * DH_;
    // K tile: 64x192 bf16 = 1536 x 16B chunks (24 per row)
    #pragma unroll
    for (int i = 0; i < 12; ++i) {
      int idx = tid + i * 128;
      int r = idx / 24, c8 = (idx % 24) * 8;
      cp16_g2l(&ks[r * QKD_ + c8], Ksrc + (long long)r * QKD_ + c8);
    }
    // V tile: 64x128 bf16 = 1024 x 16B chunks (16 per row)
    #pragma unroll
    for (int i = 0; i < 8; ++i) {
      int idx = tid + i * 128;
      int r = idx >> 4, c8 = (idx & 15) * 8;
      cp16_g2l(&vs[r * DH_ + c8], Vsrc + (long long)r * DH_ + c8);
    }
    async_copy_wait();
    __syncthreads();

    // S = Q * K^T  (16x64 strip per wave)
    v8f s[4];
    #pragma unroll
    for (int c = 0; c < 4; ++c)
      #pragma unroll
      for (int e = 0; e < 8; ++e) s[c][e] = 0.0f;
    #pragma unroll
    for (int kk = 0; kk < 6; ++kk) {
      #pragma unroll
      for (int c = 0; c < 4; ++c) {
        v16bf b;
        #pragma unroll
        for (int j = 0; j < 16; ++j)
          b[j] = u2bf(ks[(c * 16 + ln) * QKD_ + kk * 32 + frag_k(lane, j)]);
        s[c] = __builtin_amdgcn_wmma_f32_16x16x32_bf16(
            false, qf[kk], false, b, (short)0, s[c], false, false);
      }
    }

    // scale + causal mask + online softmax
    #pragma unroll
    for (int r = 0; r < 8; ++r) {
      int qr = qrow + r + 8 * hf;
      float mnew = -3.0e38f;
      #pragma unroll
      for (int c = 0; c < 4; ++c) {
        int kc = kbase + c * 16 + ln;
        float v = s[c][r] * scl;
        if (kc > qr) v = -1.0e9f;
        s[c][r] = v;
        mnew = fmaxf(mnew, v);
      }
      #pragma unroll
      for (int m = 1; m < 16; m <<= 1)
        mnew = fmaxf(mnew, __shfl_xor(mnew, m, 32));
      float mt = fmaxf(mrun[r], mnew);
      float alpha = __expf(mrun[r] - mt);
      float rsum = 0.0f;
      #pragma unroll
      for (int c = 0; c < 4; ++c) {
        float p = __expf(s[c][r] - mt);
        s[c][r] = p;
        rsum += p;
      }
      #pragma unroll
      for (int m = 1; m < 16; m <<= 1) rsum += __shfl_xor(rsum, m, 32);
      lrun[r] = lrun[r] * alpha + rsum;
      mrun[r] = mt;
      #pragma unroll
      for (int nt = 0; nt < 8; ++nt) o[nt][r] *= alpha;
    }

    // stage P strip (bf16) to wave-private LDS, re-fragment as A for P*V
    #pragma unroll
    for (int c = 0; c < 4; ++c)
      #pragma unroll
      for (int r = 0; r < 8; ++r)
        ps[wave * 1024 + (r + 8 * hf) * 64 + c * 16 + ln] = f2bf_u16(s[c][r]);

    #pragma unroll
    for (int k2 = 0; k2 < 2; ++k2) {
      v16bf a;
      #pragma unroll
      for (int j = 0; j < 16; ++j)
        a[j] = u2bf(ps[wave * 1024 + ln * 64 + k2 * 32 + frag_k(lane, j)]);
      #pragma unroll
      for (int nt = 0; nt < 8; ++nt) {
        v16bf b;   // V is [K=64][N=128] -> transposed fragment load
        #pragma unroll
        for (int j = 0; j < 16; ++j)
          b[j] = u2bf(vs[(k2 * 32 + frag_k(lane, j)) * DH_ + nt * 16 + ln]);
        o[nt] = __builtin_amdgcn_wmma_f32_16x16x32_bf16(
            false, a, false, b, (short)0, o[nt], false, false);
      }
    }
    __syncthreads();
  }

  // epilogue: normalize, write bf16 O[t, h'*NH + head]
  #pragma unroll
  for (int r = 0; r < 8; ++r) {
    float inv = 1.0f / lrun[r];
    int row = qrow + r + 8 * hf;
    #pragma unroll
    for (int nt = 0; nt < 8; ++nt) {
      int hc = nt * 16 + ln;
      Ob[(long long)row * (DH_ * NH_) + hc * NH_ + head] = f2bf_u16(o[nt][r] * inv);
    }
  }
}

// ============================================================================
extern "C" void kernel_launch(void* const* d_in, const int* in_sizes, int n_in,
                              void* d_out, int out_size, void* d_ws, size_t ws_size,
                              hipStream_t stream) {
  (void)in_sizes; (void)n_in; (void)out_size; (void)ws_size;
  const float* h    = (const float*)d_in[0];
  const float* fc   = (const float*)d_in[1];
  const float* fs   = (const float*)d_in[2];
  // d_in[3] mask (computed in-kernel), d_in[4..5] caches (fully overwritten)
  const float* w_dkv = (const float*)d_in[6];
  const float* w_uk  = (const float*)d_in[7];
  const float* w_uv  = (const float*)d_in[8];
  const float* w_dq  = (const float*)d_in[9];
  const float* w_uq  = (const float*)d_in[10];
  const float* w_qr  = (const float*)d_in[11];
  const float* w_kr  = (const float*)d_in[12];
  const float* w_o   = (const float*)d_in[13];

  // outputs: y fp32 [T,D], then ck bf16 [T,DRH], then ckv bf16 [T,DC]
  float* y = (float*)d_out;
  __hip_bfloat16* ck_out  = (__hip_bfloat16*)(y + (long long)T_ * D_);
  __hip_bfloat16* ckv_out = ck_out + (long long)T_ * DRH_;

  // ---- workspace carve-up (bf16 elements unless noted) ----
  unsigned short* h16   = (unsigned short*)d_ws;               // T*D
  unsigned short* wdkv16 = h16   + (long long)T_ * D_;         // DC*D
  unsigned short* wdq16  = wdkv16 + (long long)DC_ * D_;       // DC*D
  unsigned short* wkr16  = wdq16  + (long long)DC_ * D_;       // DRH*D
  unsigned short* wuk16  = wkr16  + (long long)DRH_ * D_;      // DH*NH*DC
  unsigned short* wuv16  = wuk16  + (long long)DH_ * NH_ * DC_;
  unsigned short* wuq16  = wuv16  + (long long)DH_ * NH_ * DC_;
  unsigned short* wqr16  = wuq16  + (long long)DH_ * NH_ * DC_; // DRH*NH*DC
  unsigned short* wo16   = wqr16  + (long long)DRH_ * NH_ * DC_; // D*DH*NH
  unsigned short* ckv16  = wo16   + (long long)D_ * DH_ * NH_; // T*DC
  unsigned short* cq16   = ckv16  + (long long)T_ * DC_;       // T*DC
  unsigned short* Qb     = cq16   + (long long)T_ * DC_;       // NH*T*192
  unsigned short* Kb     = Qb + (long long)NH_ * T_ * QKD_;    // NH*T*192
  unsigned short* Vb     = Kb + (long long)NH_ * T_ * QKD_;    // NH*T*128
  unsigned short* Ob     = Vb + (long long)NH_ * T_ * DH_;     // T*2048
  float* kr_ws = (float*)(Ob + (long long)T_ * D_);            // T*DRH fp32

  dim3 blk(128);
  // ---- one-time bf16 packing of h and weights ----
  struct { const float* src; unsigned short* dst; int n; } packs[] = {
    { h,     h16,    T_ * D_ },
    { w_dkv, wdkv16, DC_ * D_ },
    { w_dq,  wdq16,  DC_ * D_ },
    { w_kr,  wkr16,  DRH_ * D_ },
    { w_uk,  wuk16,  DH_ * NH_ * DC_ },
    { w_uv,  wuv16,  DH_ * NH_ * DC_ },
    { w_uq,  wuq16,  DH_ * NH_ * DC_ },
    { w_qr,  wqr16,  DRH_ * NH_ * DC_ },
    { w_o,   wo16,   D_ * DH_ * NH_ },
  };
  for (int p = 0; p < 9; ++p) {
    int n4 = packs[p].n / 4;
    pack_bf16_kernel<<<(n4 + 255) / 256, 256, 0, stream>>>(
        packs[p].src, packs[p].dst, n4);
  }

  // ---- down-projections: [T,D] x W[N,D]^T ----
  wmma_gemm_bf16<<<dim3(DC_/64, T_/64, 1), blk, 0, stream>>>(
      h16, D_, 0, wdkv16, D_, 0, ckv16, DC_, 0, D_, 1);
  wmma_gemm_bf16<<<dim3(DC_/64, T_/64, 1), blk, 0, stream>>>(
      h16, D_, 0, wdq16, D_, 0, cq16, DC_, 0, D_, 1);
  wmma_gemm_bf16<<<dim3(DRH_/64, T_/64, 1), blk, 0, stream>>>(
      h16, D_, 0, wkr16, D_, 0, kr_ws, DRH_, 0, D_, 0);

  // ---- caches ----
  (void)hipMemcpyAsync(ckv_out, ckv16, (size_t)T_ * DC_ * sizeof(unsigned short),
                       hipMemcpyDeviceToDevice, stream);
  rope_k_kernel<<<(T_ * 32 + 255) / 256, 256, 0, stream>>>(
      kr_ws, fc, fs, ck_out, Kb);

  // ---- up-projections, batched over heads (grid.z = NH) ----
  // per-head weight row h' at W + (h'*NH + n)*DC  ->  ldb = NH*DC, strideB = DC
  wmma_gemm_bf16<<<dim3(DH_/64, T_/64, NH_), blk, 0, stream>>>(
      ckv16, DC_, 0, wuk16, NH_ * DC_, DC_, Kb, QKD_, (long long)T_ * QKD_, DC_, 1);
  wmma_gemm_bf16<<<dim3(DH_/64, T_/64, NH_), blk, 0, stream>>>(
      ckv16, DC_, 0, wuv16, NH_ * DC_, DC_, Vb, DH_, (long long)T_ * DH_, DC_, 1);
  wmma_gemm_bf16<<<dim3(DH_/64, T_/64, NH_), blk, 0, stream>>>(
      cq16, DC_, 0, wuq16, NH_ * DC_, DC_, Qb, QKD_, (long long)T_ * QKD_, DC_, 1);
  wmma_gemm_bf16<<<dim3(DRH_/64, T_/64, NH_), blk, 0, stream>>>(
      cq16, DC_, 0, wqr16, NH_ * DC_, DC_, Qb + DH_, QKD_, (long long)T_ * QKD_, DC_, 1);

  rope_q_kernel<<<(NH_ * T_ * 32 + 255) / 256, 256, 0, stream>>>(Qb, fc, fs);

  // ---- attention ----
  mla_attn_kernel<<<dim3(T_/64, NH_), blk, 0, stream>>>(Qb, Kb, Vb, Ob);

  // ---- output projection: y[T,D] = O[T,2048] x w_o[D,2048]^T (fp32 out) ----
  wmma_gemm_bf16<<<dim3(D_/64, T_/64, 1), blk, 0, stream>>>(
      Ob, DH_ * NH_, 0, wo16, DH_ * NH_, 0, y, D_, 0, DH_ * NH_, 0);
}
